// MultiHeadRelationalModule_60181081752248
// MI455X (gfx1250) — matrix operand order repaired and checked by reading, not compiled, not run on hardware
//
#include <hip/hip_runtime.h>
#include <hip/hip_bf16.h>
#include <math.h>

// ---------------------------------------------------------------------------
// MI455X (gfx1250) implementation: all heavy GEMMs via v_wmma_f32_16x16x32_bf16
// (bf16 inputs, f32 accumulate). Scalar math (LN, softmax, elu) stays f32.
// ---------------------------------------------------------------------------

typedef __attribute__((ext_vector_type(16))) __bf16 v16bf;
typedef __attribute__((ext_vector_type(8)))  float  v8f;
typedef __attribute__((ext_vector_type(4)))  unsigned int v4u;

union FragU { v4u q[2]; v16bf v; unsigned short s[16]; };

static constexpr int NB   = 64;    // batch
static constexpr int NH   = 4;     // heads
static constexpr int NPIX = 596;   // pixels
static constexpr int PPAD = 608;   // 38*16 padded pixels
static constexpr int DH   = 64;    // head dim
static constexpr int FEAT = 34;    // conv feats (+coords)
static constexpr int CP   = 256;   // proj width
static constexpr int RT   = 38;    // row tiles of 16 over PPAD
static constexpr int NT_W1 = 38;   // G1 B: 128 x 608 -> 4 ktiles x 38 ntiles
static constexpr int KT_AL = 19;   // G2 B: 608 x 608 -> 19 ktiles x 38 ntiles
static constexpr int NT_AL = 38;

__device__ inline unsigned short f2bf(float f) {
    union { float f; unsigned u; } c; c.f = f;
    unsigned u = c.u;
    unsigned r = ((u >> 16) & 1u) + 0x7fffu;
    return (unsigned short)((u + r) >> 16);
}

// A-fragment (16x32 bf16) from a row-major matrix: lane m holds row m,
// K contiguous {h*16..h*16+15} (matches B packing k = 16h + j).
__device__ inline v16bf load_afrag(const unsigned short* base, int stride, int k0, int lane) {
    int half = (lane >> 4) & 1;
    int m    = lane & 15;
    const unsigned short* p0 = base + (size_t)m * stride + k0 + half * 16;
    FragU u;
    u.q[0] = *reinterpret_cast<const v4u*>(p0);
    u.q[1] = *reinterpret_cast<const v4u*>(p0 + 8);
    return u.v;
}

// B-fragment (32x16 bf16) from pre-packed buffer: tile-major, 32 lanes x 16 elems.
__device__ inline v16bf load_pfrag(const unsigned short* packed, int tile, int lane) {
    const v4u* p = reinterpret_cast<const v4u*>(packed + ((size_t)tile * 32 + lane) * 16);
    FragU u;
    u.q[0] = p[0];
    u.q[1] = p[1];
    return u.v;
}

__device__ inline v8f wmma_bf(v16bf a, v16bf b, v8f c) {
    return __builtin_amdgcn_wmma_f32_16x16x32_bf16(false, a, false, b, (short)0, c, false, false);
}

__device__ inline float elu1(float x) { return x > 0.f ? x : (expf(x) - 1.f); }

// ---------------------------------------------------------------------------
// Kernel 0: pack B-operand weights into fragment order (bf16), zero-padded.
// B packing: within a 32x16 tile, lane = (k_local>>4)*16 + n, element j = k_local&15.
// ---------------------------------------------------------------------------
__global__ void mhr_prep_pack(const float* qlin_w, const float* klin_w,
                              const float* alin_w, const float* lin1_w,
                              unsigned short* W1f, unsigned short* ALf,
                              unsigned short* L1f) {
    int idx = blockIdx.x * 256 + threadIdx.x;
    const int nW1 = 4 * NT_W1 * 512;       // 77824
    const int nAL = KT_AL * NT_AL * 512;   // 369664
    const int nL1 = 8 * 4 * 512;           // 16384
    if (idx < nW1) {
        int tl = idx >> 9, e = idx & 511;
        int lane = e >> 4, j = e & 15;
        int kt = tl / NT_W1, nt = tl % NT_W1;
        int half = lane >> 4, n = lane & 15;
        int k = kt * 32 + half * 16 + j;
        int col = nt * 16 + n;
        float v = 0.f;
        if (col < NPIX) v = (k < 64) ? qlin_w[k * NPIX + col] : klin_w[(k - 64) * NPIX + col];
        W1f[idx] = f2bf(v);
        return;
    }
    idx -= nW1;
    if (idx < nAL) {
        int tl = idx >> 9, e = idx & 511;
        int lane = e >> 4, j = e & 15;
        int kt = tl / NT_AL, nt = tl % NT_AL;
        int half = lane >> 4, n = lane & 15;
        int k = kt * 32 + half * 16 + j;
        int col = nt * 16 + n;
        float v = (k < NPIX && col < NPIX) ? alin_w[(size_t)k * NPIX + col] : 0.f;
        ALf[idx] = f2bf(v);
        return;
    }
    idx -= nAL;
    if (idx < nL1) {
        int tl = idx >> 9, e = idx & 511;
        int lane = e >> 4, j = e & 15;
        int kt = tl / 4, nt = tl % 4;
        int half = lane >> 4, n = lane & 15;
        int k = kt * 32 + half * 16 + j;   // < 256
        int col = nt * 16 + n;             // < 64
        L1f[idx] = f2bf(lin1_w[k * 64 + col]);
    }
}

// ---------------------------------------------------------------------------
// conv1 (4->16, 2x2 VALID) + relu : out (64,16,150,5)
// ---------------------------------------------------------------------------
__global__ void mhr_conv1(const float* x, const float* w, const float* bias, float* out) {
    int idx = blockIdx.x * 256 + threadIdx.x;
    if (idx >= NB * 16 * 150 * 5) return;
    int j = idx % 5, t = idx / 5;
    int i = t % 150; t /= 150;
    int oc = t % 16; int b = t / 16;
    float acc = bias[oc];
    for (int ic = 0; ic < 4; ++ic)
        for (int di = 0; di < 2; ++di)
            for (int dj = 0; dj < 2; ++dj)
                acc += x[(((size_t)b * 4 + ic) * 151 + i + di) * 6 + j + dj] *
                       w[((oc * 4 + ic) * 2 + di) * 2 + dj];
    out[idx] = acc > 0.f ? acc : 0.f;
}

// ---------------------------------------------------------------------------
// conv2 (16->32, 2x2 VALID) + relu + coords -> feats (64, 596, 34)
// ---------------------------------------------------------------------------
__global__ void mhr_conv2_feats(const float* c1, const float* w, const float* bias, float* feats) {
    int idx = blockIdx.x * 256 + threadIdx.x;
    if (idx >= NB * NPIX * FEAT) return;
    int c = idx % FEAT, t = idx / FEAT;
    int p = t % NPIX; int b = t / NPIX;
    int i = p >> 2, j = p & 3;  // i<149, j<4
    float v;
    if (c < 32) {
        float acc = bias[c];
        for (int ic = 0; ic < 16; ++ic)
            for (int di = 0; di < 2; ++di)
                for (int dj = 0; dj < 2; ++dj)
                    acc += c1[(((size_t)b * 16 + ic) * 150 + i + di) * 5 + j + dj] *
                           w[((c * 16 + ic) * 2 + di) * 2 + dj];
        v = acc > 0.f ? acc : 0.f;
    } else if (c == 32) {
        v = (float)j / 4.0f;
    } else {
        v = (float)i / 149.0f;
    }
    feats[idx] = v;
}

// ---------------------------------------------------------------------------
// K/Q/V projections: feats(596x34) @ w(34x256) + b -> Pbuf[t][b][p][256] (f32),
// plus deterministic per-block sum/sumsq partials for the per-sample LN.
// block = ((t*64+b)*19 + ptile), 256 threads (thread = output col).
// ---------------------------------------------------------------------------
__global__ __launch_bounds__(256) void mhr_proj(
        const float* feats,
        const float* kw, const float* kb, const float* qw, const float* qb,
        const float* vw, const float* vb,
        float* Pbuf, float* partials) {
    __shared__ float ft[32][FEAT];
    __shared__ float rs[256], rs2[256];
    int blk = blockIdx.x;
    int pt = blk % 19; int tb = blk / 19;
    int b = tb % NB; int t = tb / NB;
    const float* W = (t == 0) ? kw : (t == 1) ? qw : vw;
    const float* B = (t == 0) ? kb : (t == 1) ? qb : vb;
    for (int i = threadIdx.x; i < 32 * FEAT; i += 256) {
        int pp = i / FEAT, cc = i % FEAT;
        int p = pt * 32 + pp;
        ft[pp][cc] = (p < NPIX) ? feats[((size_t)b * NPIX + p) * FEAT + cc] : 0.f;
    }
    __syncthreads();
    int c = threadIdx.x;
    float s = 0.f, s2 = 0.f;
    for (int pp = 0; pp < 32; ++pp) {
        int p = pt * 32 + pp;
        if (p >= NPIX) break;
        float acc = B[c];
        for (int cc = 0; cc < FEAT; ++cc) acc += ft[pp][cc] * W[cc * CP + c];
        Pbuf[(((size_t)t * NB + b) * NPIX + p) * CP + c] = acc;
        s += acc; s2 += acc * acc;
    }
    rs[c] = s; rs2[c] = s2;
    __syncthreads();
    for (int st = 128; st > 0; st >>= 1) {
        if (c < st) { rs[c] += rs[c + st]; rs2[c] += rs2[c + st]; }
        __syncthreads();
    }
    if (c == 0) {
        partials[(size_t)blk * 2 + 0] = rs[0];
        partials[(size_t)blk * 2 + 1] = rs2[0];
    }
}

// Combine 19 partials per (t,b) -> mu, rstd.
__global__ void mhr_stats(const float* partials, float* stats) {
    if (threadIdx.x != 0) return;
    int id = blockIdx.x;  // t*64+b, 192 blocks
    double s = 0.0, s2 = 0.0;
    const float* p = partials + (size_t)id * 19 * 2;
    for (int i = 0; i < 19; ++i) { s += p[i * 2]; s2 += p[i * 2 + 1]; }
    double n = (double)NH * NPIX * DH;
    double mu = s / n;
    double var = s2 / n - mu * mu;
    stats[id * 2 + 0] = (float)mu;
    stats[id * 2 + 1] = rsqrtf((float)var + 1e-5f);
}

// ---------------------------------------------------------------------------
// LN apply: normalize + affine, convert to bf16, write:
//   t=1 (Q) -> QK cols 0..63, t=0 (K) -> QK cols 64..127 (row-major, 608 rows)
//   t=2 (V) -> B-fragment-packed Vfrag per (b,h), zero-padded rows.
// ---------------------------------------------------------------------------
__global__ void mhr_lnapply(const float* Pbuf, const float* stats,
                            const float* kg, const float* kbta,
                            const float* qg, const float* qbta,
                            const float* vg, const float* vbta,
                            unsigned short* QK, unsigned short* Vfrag) {
    int idx = blockIdx.x * 256 + threadIdx.x;  // < 3*64*4*608*64
    int d = idx & 63; int r = idx >> 6;
    int p = r % PPAD; r /= PPAD;
    int h = r & 3; r >>= 2;
    int b = r % NB; int t = r / NB;
    float val = 0.f;
    if (p < NPIX) {
        float x = Pbuf[(((size_t)t * NB + b) * NPIX + p) * CP + h * DH + d];
        float mu = stats[(t * NB + b) * 2 + 0];
        float rstd = stats[(t * NB + b) * 2 + 1];
        const float* g  = (t == 0) ? kg : (t == 1) ? qg : vg;
        const float* be = (t == 0) ? kbta : (t == 1) ? qbta : vbta;
        int gi = (h * NPIX + p) * DH + d;
        val = (x - mu) * rstd * g[gi] + be[gi];
    }
    unsigned short o = f2bf(val);
    if (t < 2) {
        QK[(((size_t)b * NH + h) * PPAD + p) * 128 + (t == 1 ? 0 : 64) + d] = o;
    } else {
        int bh = b * NH + h;
        int kt = p >> 5, kl = p & 31;
        int half = kl >> 4, j = kl & 15;
        int lane = half * 16 + (d & 15);
        int nt = d >> 4;
        Vfrag[((((size_t)bh * KT_AL + kt) * 4 + nt) * 512) + lane * 16 + j] = o;
    }
}

// ---------------------------------------------------------------------------
// G1: A_pre = elu([Q|K](608x128) @ W1f(128x608) + qlin_b + klin_b), bf16,
// pad cols zeroed. block = (bh, row-tile), 4 waves split N-tiles.
// ---------------------------------------------------------------------------
__global__ __launch_bounds__(128) void mhr_g1(const unsigned short* QK,
                                              const unsigned short* W1f,
                                              const float* qlinb, const float* klinb,
                                              unsigned short* Abuf) {
    int blk = blockIdx.x;
    int rt = blk % RT; int bh = blk / RT;
    const unsigned short* Arow0 = QK + ((size_t)bh * PPAD + rt * 16) * 128;
    unsigned short* outb = Abuf + ((size_t)bh * PPAD + rt * 16) * PPAD;
    int lane = threadIdx.x & 31, w = threadIdx.x >> 5;
    for (int nt = w; nt < NT_W1; nt += 4) {
        v8f acc = {0.f, 0.f, 0.f, 0.f, 0.f, 0.f, 0.f, 0.f};
        for (int ks = 0; ks < 4; ++ks) {
            v16bf a = load_afrag(Arow0, 128, ks * 32, lane);
            v16bf bb = load_pfrag(W1f, ks * NT_W1 + nt, lane);
            acc = wmma_bf(a, bb, acc);
        }
        int m0 = (lane >> 4) << 3, n = lane & 15;
        int col = nt * 16 + n;
        for (int r = 0; r < 8; ++r) {
            float o = 0.f;
            if (col < NPIX) o = elu1(acc[r] + qlinb[col] + klinb[col]);
            outb[(size_t)(m0 + r) * PPAD + col] = f2bf(o);
        }
    }
}

// ---------------------------------------------------------------------------
// G2 + softmax fused: Z = A_pre(16x608) @ alin(608x608) + alin_b; row softmax;
// overwrite A_pre in place with bf16 A. A rows staged in LDS; Z tile in LDS.
// ---------------------------------------------------------------------------
__global__ __launch_bounds__(128) void mhr_g2_softmax(unsigned short* Abuf,
                                                      const unsigned short* ALf,
                                                      const float* alinb) {
    __shared__ __align__(16) unsigned short Asub[16 * PPAD];
    __shared__ float Zs[16 * PPAD];
    __shared__ float red[128];
    __shared__ float rowmax[16], rowinv[16];
    int blk = blockIdx.x;
    int rt = blk % RT; int bh = blk / RT;
    unsigned short* Arows = Abuf + ((size_t)bh * PPAD + rt * 16) * PPAD;
    {   // stage 16 contiguous rows (16*608 bf16) into LDS
        const v4u* src = reinterpret_cast<const v4u*>(Arows);
        v4u* dst = reinterpret_cast<v4u*>(Asub);
        for (int i = threadIdx.x; i < 16 * PPAD / 8; i += 128) dst[i] = src[i];
    }
    __syncthreads();
    int lane = threadIdx.x & 31, w = threadIdx.x >> 5;
    for (int nt = w; nt < NT_AL; nt += 4) {
        v8f acc = {0.f, 0.f, 0.f, 0.f, 0.f, 0.f, 0.f, 0.f};
        for (int ks = 0; ks < KT_AL; ++ks) {
            if (ks + 1 < KT_AL)
                __builtin_prefetch(ALf + ((size_t)((ks + 1) * NT_AL + nt) * 512), 0, 0);
            v16bf a = load_afrag(Asub, PPAD, ks * 32, lane);
            v16bf bb = load_pfrag(ALf, ks * NT_AL + nt, lane);
            acc = wmma_bf(a, bb, acc);
        }
        int m0 = (lane >> 4) << 3, n = lane & 15;
        for (int r = 0; r < 8; ++r) Zs[(size_t)(m0 + r) * PPAD + nt * 16 + n] = acc[r];
    }
    __syncthreads();
    int row = threadIdx.x >> 3, s = threadIdx.x & 7;
    float mx = -3.0e38f;
    for (int c = s; c < NPIX; c += 8) mx = fmaxf(mx, Zs[row * PPAD + c] + alinb[c]);
    red[threadIdx.x] = mx;
    __syncthreads();
    if (s == 0) {
        float m2 = red[row * 8];
        for (int i = 1; i < 8; ++i) m2 = fmaxf(m2, red[row * 8 + i]);
        rowmax[row] = m2;
    }
    __syncthreads();
    float rm = rowmax[row], sm = 0.f;
    for (int c = s; c < NPIX; c += 8) sm += expf(Zs[row * PPAD + c] + alinb[c] - rm);
    red[threadIdx.x] = sm;
    __syncthreads();
    if (s == 0) {
        float tsum = 0.f;
        for (int i = 0; i < 8; ++i) tsum += red[row * 8 + i];
        rowinv[row] = 1.0f / tsum;
    }
    __syncthreads();
    for (int e = threadIdx.x; e < 16 * PPAD; e += 128) {
        int rr = e / PPAD, c = e - rr * PPAD;
        float v = 0.f;
        if (c < NPIX) v = expf(Zs[e] + alinb[c] - rowmax[rr]) * rowinv[rr];
        Arows[e] = f2bf(v);
    }
}

// ---------------------------------------------------------------------------
// G3: E = A(608x608) @ V(608x64); write E (b, p, h*64+d) f32 for p<596.
// ---------------------------------------------------------------------------
__global__ __launch_bounds__(128) void mhr_g3(const unsigned short* Abuf,
                                              const unsigned short* Vfrag,
                                              float* Ebuf) {
    int blk = blockIdx.x;
    int rt = blk % RT; int bh = blk / RT;
    int b = bh >> 2, h = bh & 3;
    const unsigned short* Arow0 = Abuf + ((size_t)bh * PPAD + rt * 16) * PPAD;
    const unsigned short* Vb = Vfrag + (size_t)bh * KT_AL * 4 * 512;
    int lane = threadIdx.x & 31, nt = threadIdx.x >> 5;  // nt < 4
    v8f acc = {0.f, 0.f, 0.f, 0.f, 0.f, 0.f, 0.f, 0.f};
    for (int ks = 0; ks < KT_AL; ++ks) {
        if (ks + 1 < KT_AL)
            __builtin_prefetch(Vb + ((size_t)((ks + 1) * 4 + nt) * 512), 0, 0);
        v16bf a = load_afrag(Arow0, PPAD, ks * 32, lane);
        v16bf bb = load_pfrag(Vb, ks * 4 + nt, lane);
        acc = wmma_bf(a, bb, acc);
    }
    int m0 = (lane >> 4) << 3, n = lane & 15;
    for (int r = 0; r < 8; ++r) {
        int p = rt * 16 + m0 + r;
        if (p < NPIX)
            Ebuf[((size_t)b * NPIX + p) * CP + h * DH + nt * 16 + n] = acc[r];
    }
}

// ---------------------------------------------------------------------------
// lin1: relu(E(596x256) @ lin1_w(256x64) + b); A-frags built on the fly (f32->bf16).
// ---------------------------------------------------------------------------
__global__ __launch_bounds__(128) void mhr_lin1(const float* Ebuf,
                                                const unsigned short* L1f,
                                                const float* l1b, float* L1out) {
    int blk = blockIdx.x;
    int rt = blk % RT; int b = blk / RT;
    int lane = threadIdx.x & 31, nt = threadIdx.x >> 5;  // nt < 4
    int half = (lane >> 4) & 1, m = lane & 15;
    int p = rt * 16 + m;
    v8f acc = {0.f, 0.f, 0.f, 0.f, 0.f, 0.f, 0.f, 0.f};
    for (int ks = 0; ks < 8; ++ks) {
        FragU u;
        if (p < NPIX) {
            const float* src = Ebuf + ((size_t)b * NPIX + p) * CP;
            for (int j = 0; j < 16; ++j) {
                int k = ks * 32 + half * 16 + j;
                u.s[j] = f2bf(src[k]);
            }
        } else {
            u.q[0] = (v4u){0u, 0u, 0u, 0u};
            u.q[1] = (v4u){0u, 0u, 0u, 0u};
        }
        v16bf bb = load_pfrag(L1f, ks * 4 + nt, lane);
        acc = wmma_bf(u.v, bb, acc);
    }
    int m0 = (lane >> 4) << 3, n = lane & 15;
    int col = nt * 16 + n;
    for (int r = 0; r < 8; ++r) {
        int pr = rt * 16 + m0 + r;
        if (pr < NPIX) {
            float v = acc[r] + l1b[col];
            L1out[((size_t)b * NPIX + pr) * DH + col] = v > 0.f ? v : 0.f;
        }
    }
}

// ---------------------------------------------------------------------------
// Final: LN(596x64, no affine) -> max over pixels -> lin2 -> elu. One block / b.
// max_p (x-mu)*rstd == (max_p x - mu)*rstd since rstd > 0.
// ---------------------------------------------------------------------------
__global__ __launch_bounds__(256) void mhr_final(const float* L1, const float* w2,
                                                 const float* b2, float* out) {
    __shared__ float rs[256], rs2[256];
    __shared__ float Ef[64];
    __shared__ float mus[2];
    int b = blockIdx.x, tid = threadIdx.x;
    const float* Lb = L1 + (size_t)b * NPIX * DH;
    float s = 0.f, s2 = 0.f;
    for (int i = tid; i < NPIX * DH; i += 256) {
        float v = Lb[i];
        s += v; s2 += v * v;
    }
    rs[tid] = s; rs2[tid] = s2;
    __syncthreads();
    for (int st = 128; st > 0; st >>= 1) {
        if (tid < st) { rs[tid] += rs[tid + st]; rs2[tid] += rs2[tid + st]; }
        __syncthreads();
    }
    if (tid == 0) {
        float n = (float)(NPIX * DH);
        float mu = rs[0] / n;
        float var = rs2[0] / n - mu * mu;
        mus[0] = mu;
        mus[1] = rsqrtf(var + 1e-5f);
    }
    __syncthreads();
    if (tid < DH) {
        float m = -3.0e38f;
        for (int p = 0; p < NPIX; ++p) m = fmaxf(m, Lb[(size_t)p * DH + tid]);
        Ef[tid] = (m - mus[0]) * mus[1];
    }
    __syncthreads();
    if (tid < 10) {
        float a = b2[tid];
        for (int d = 0; d < DH; ++d) a += Ef[d] * w2[d * 10 + tid];
        out[b * 10 + tid] = a > 0.f ? a : (expf(a) - 1.f);
    }
}

// ---------------------------------------------------------------------------
extern "C" void kernel_launch(void* const* d_in, const int* in_sizes, int n_in,
                              void* d_out, int out_size, void* d_ws, size_t ws_size,
                              hipStream_t stream) {
    (void)in_sizes; (void)n_in; (void)out_size; (void)ws_size;
    const float* x       = (const float*)d_in[0];
    const float* conv1_w = (const float*)d_in[1];
    const float* conv1_b = (const float*)d_in[2];
    const float* conv2_w = (const float*)d_in[3];
    const float* conv2_b = (const float*)d_in[4];
    const float* kp_w    = (const float*)d_in[5];
    const float* kp_b    = (const float*)d_in[6];
    const float* qp_w    = (const float*)d_in[7];
    const float* qp_b    = (const float*)d_in[8];
    const float* vp_w    = (const float*)d_in[9];
    const float* vp_b    = (const float*)d_in[10];
    const float* klin_w  = (const float*)d_in[11];
    const float* klin_b  = (const float*)d_in[12];
    const float* qlin_w  = (const float*)d_in[13];
    const float* qlin_b  = (const float*)d_in[14];
    const float* alin_w  = (const float*)d_in[15];
    const float* alin_b  = (const float*)d_in[16];
    const float* knorm_g = (const float*)d_in[17];
    const float* knorm_b = (const float*)d_in[18];
    const float* qnorm_g = (const float*)d_in[19];
    const float* qnorm_b = (const float*)d_in[20];
    const float* vnorm_g = (const float*)d_in[21];
    const float* vnorm_b = (const float*)d_in[22];
    const float* lin1_w  = (const float*)d_in[23];
    const float* lin1_b  = (const float*)d_in[24];
    const float* lin2_w  = (const float*)d_in[25];
    const float* lin2_b  = (const float*)d_in[26];
    float* out = (float*)d_out;
    char* ws = (char*)d_ws;

    // Workspace layout (256B aligned offsets), ~375 MB total.
    const size_t o_conv1 = 0;                            // 3,072,000
    const size_t o_feats = o_conv1 + 3072000;            // 5,187,584
    const size_t o_P     = o_feats + 5187584;            // 117,178,368 (reused: E, L1)
    const size_t o_part  = o_P + 117178368;              // 29,184
    const size_t o_stats = o_part + 29184;               // 1,536
    const size_t o_QK    = o_stats + 1536;               // 39,845,888
    const size_t o_V     = o_QK + 39845888;              // 19,922,944
    const size_t o_W1    = o_V + 19922944;               // 155,648
    const size_t o_alin  = o_W1 + 155648;                // 739,328
    const size_t o_l1w   = o_alin + 739328;              // 32,768
    const size_t o_A     = o_l1w + 32768;                // 189,267,968

    float* conv1buf = (float*)(ws + o_conv1);
    float* featsbuf = (float*)(ws + o_feats);
    float* Pbuf     = (float*)(ws + o_P);
    float* Ebuf     = (float*)(ws + o_P);                 // alias (after LN consumes P)
    float* L1buf    = (float*)(ws + o_P + 39059456);      // alias tail of P region
    float* partials = (float*)(ws + o_part);
    float* stats    = (float*)(ws + o_stats);
    unsigned short* QKbuf = (unsigned short*)(ws + o_QK);
    unsigned short* Vfrag = (unsigned short*)(ws + o_V);
    unsigned short* W1f   = (unsigned short*)(ws + o_W1);
    unsigned short* ALf   = (unsigned short*)(ws + o_alin);
    unsigned short* L1f   = (unsigned short*)(ws + o_l1w);
    unsigned short* Abuf  = (unsigned short*)(ws + o_A);

    mhr_prep_pack<<<1812, 256, 0, stream>>>(qlin_w, klin_w, alin_w, lin1_w, W1f, ALf, L1f);
    mhr_conv1<<<3000, 256, 0, stream>>>(x, conv1_w, conv1_b, conv1buf);
    mhr_conv2_feats<<<5066, 256, 0, stream>>>(conv1buf, conv2_w, conv2_b, featsbuf);
    mhr_proj<<<3 * NB * 19, 256, 0, stream>>>(featsbuf, kp_w, kp_b, qp_w, qp_b, vp_w, vp_b,
                                              Pbuf, partials);
    mhr_stats<<<3 * NB, 32, 0, stream>>>(partials, stats);
    mhr_lnapply<<<116736, 256, 0, stream>>>(Pbuf, stats, knorm_g, knorm_b, qnorm_g, qnorm_b,
                                            vnorm_g, vnorm_b, QKbuf, Vfrag);
    mhr_g1<<<NB * NH * RT, 128, 0, stream>>>(QKbuf, W1f, qlin_b, klin_b, Abuf);
    mhr_g2_softmax<<<NB * NH * RT, 128, 0, stream>>>(Abuf, ALf, alin_b);
    mhr_g3<<<NB * NH * RT, 128, 0, stream>>>(Abuf, Vfrag, Ebuf);
    mhr_lin1<<<NB * RT, 128, 0, stream>>>(Ebuf, L1f, lin1_b, L1buf);
    mhr_final<<<NB, 256, 0, stream>>>(L1buf, lin2_w, lin2_b, out);
}
